// GroupedQueryAttention_48730698940435
// MI455X (gfx1250) — 16-bit and 8-bit matrix operand order repaired and checked by reading, compiled, not run on hardware
//
#include <hip/hip_runtime.h>
#include <cstdint>
#include <cstddef>

#define B_ 2
#define S_ 1024
#define D_ 4096
#define H_ 32
#define KV_ 8
#define HD_ 128
#define GROUP_ (H_ / KV_)

typedef __attribute__((ext_vector_type(8)))  int      v8i;
typedef __attribute__((ext_vector_type(8)))  float    v8f;
typedef __attribute__((ext_vector_type(16))) _Float16 v16h;

__device__ __forceinline__ void async_copy_b128(uint32_t lds_off,
                                                const void* sbase,
                                                int voffset) {
  asm volatile("global_load_async_to_lds_b128 %0, %1, %2"
               :
               : "v"(lds_off), "v"(voffset), "s"(sbase)
               : "memory");
}

__device__ __forceinline__ void wait_async0() {
  asm volatile("s_wait_asynccnt 0x0" ::: "memory");
}

__global__ __launch_bounds__(256) void quant_rows_kernel(
    const float* __restrict__ x, int8_t* __restrict__ xq,
    float* __restrict__ sx, int K) {
  __shared__ float red[256];
  const int row = blockIdx.x;
  const int tid = threadIdx.x;
  const float* xr = x + (size_t)row * K;
  float amax = 0.0f;
  for (int i = tid; i < K; i += 256) amax = fmaxf(amax, fabsf(xr[i]));
  red[tid] = amax;
  __syncthreads();
  for (int s = 128; s > 0; s >>= 1) {
    if (tid < s) red[tid] = fmaxf(red[tid], red[tid + s]);
    __syncthreads();
  }
  float scale = red[0] / 127.0f;
  if (scale == 0.0f) scale = 1.0f;
  if (tid == 0) sx[row] = scale;
  const float inv = 1.0f / scale;
  int8_t* qr = xq + (size_t)row * K;
  for (int i = tid; i < K; i += 256) {
    float v = rintf(xr[i] * inv);
    v = fminf(fmaxf(v, -127.0f), 127.0f);
    qr[i] = (int8_t)v;
  }
}

__global__ __launch_bounds__(256) void gemm_i8_kernel(
    const int8_t* __restrict__ xq, const float* __restrict__ sx,
    const int8_t* __restrict__ w, const float* __restrict__ wsc,
    const float* __restrict__ bias, float* __restrict__ y,
    int M, int N, int K) {
  __shared__ int8_t As[2][128 * 64];
  __shared__ int8_t Bs[2][128 * 64];
  const int tid = threadIdx.x;
  const int wave = tid >> 5;
  const int lane = tid & 31;
  const int ln = lane & 15;
  const int hl = lane >> 4;
  const int m0 = blockIdx.y * 128;
  const int n0 = blockIdx.x * 128;
  const int mw = wave >> 1;
  const int nw = wave & 1;

  v8i acc[2][4];
  for (int mi = 0; mi < 2; ++mi)
    for (int ni = 0; ni < 4; ++ni)
      for (int e = 0; e < 8; ++e) acc[mi][ni][e] = 0;

  const int ldr = tid >> 1;
  const int ldc = (tid & 1) * 32;
  const int offA = (m0 + ldr) * K + ldc;
  const int offB = (n0 + ldr) * K + ldc;
  const uint32_t ldsA[2] = {
      (uint32_t)(uintptr_t)(&As[0][ldr * 64 + ldc]),
      (uint32_t)(uintptr_t)(&As[1][ldr * 64 + ldc])};
  const uint32_t ldsB[2] = {
      (uint32_t)(uintptr_t)(&Bs[0][ldr * 64 + ldc]),
      (uint32_t)(uintptr_t)(&Bs[1][ldr * 64 + ldc])};

  async_copy_b128(ldsA[0], xq, offA);
  async_copy_b128(ldsA[0] + 16, xq, offA + 16);
  async_copy_b128(ldsB[0], w, offB);
  async_copy_b128(ldsB[0] + 16, w, offB + 16);

  for (int k0 = 0; k0 < K; k0 += 64) {
    wait_async0();
    __syncthreads();

    const int buf = (k0 >> 6) & 1;
    if (k0 + 64 < K) {
      const int nb = buf ^ 1;
      const int kn = k0 + 64;
      async_copy_b128(ldsA[nb], xq, offA + kn);
      async_copy_b128(ldsA[nb] + 16, xq, offA + kn + 16);
      async_copy_b128(ldsB[nb], w, offB + kn);
      async_copy_b128(ldsB[nb] + 16, w, offB + kn + 16);
    }

    v8i af[2], bf[4];
    for (int mi = 0; mi < 2; ++mi) {
      const int* Ar = (const int*)(As[buf] + (mw * 32 + mi * 16 + ln) * 64);
#pragma unroll
      for (int j = 0; j < 8; ++j)
        af[mi][j] = Ar[(j >> 1) * 4 + hl * 2 + (j & 1)];
    }
    for (int ni = 0; ni < 4; ++ni) {
      const int* Br = (const int*)(Bs[buf] + (nw * 64 + ni * 16 + ln) * 64);
#pragma unroll
      for (int j = 0; j < 8; ++j)
        bf[ni][j] = Br[(j >> 1) * 4 + hl * 2 + (j & 1)];
    }
#pragma unroll
    for (int mi = 0; mi < 2; ++mi)
#pragma unroll
      for (int ni = 0; ni < 4; ++ni)
        acc[mi][ni] = __builtin_amdgcn_wmma_i32_16x16x64_iu8(
            true, af[mi], true, bf[ni], acc[mi][ni], false, false);
  }

  for (int mi = 0; mi < 2; ++mi)
    for (int ni = 0; ni < 4; ++ni) {
      const int mt = m0 + mw * 32 + mi * 16;
      const int col = n0 + nw * 64 + ni * 16 + ln;
      const float wc = wsc[col];
      const float bb = bias ? bias[col] : 0.0f;
#pragma unroll
      for (int r = 0; r < 8; ++r) {
        const int row = mt + r + 8 * hl;
        y[(size_t)row * N + col] = (float)acc[mi][ni][r] * sx[row] * wc + bb;
      }
    }
}

__global__ __launch_bounds__(HD_) void rope_cvt_q_kernel(
    const float* __restrict__ qf, const float* __restrict__ cosT,
    const float* __restrict__ sinT, _Float16* __restrict__ qh) {
  const int s = blockIdx.x, h = blockIdx.y, b = blockIdx.z, d = threadIdx.x;
  const float* row = qf + (((size_t)(b * S_ + s)) * H_ + h) * HD_;
  const float t = row[d];
  const float rot = (d < HD_ / 2) ? -row[d + HD_ / 2] : row[d - HD_ / 2];
  float val = t * cosT[s * HD_ + d] + rot * sinT[s * HD_ + d];
  val *= 0.088388347648318447f;
  qh[(((size_t)(b * H_ + h)) * S_ + s) * HD_ + d] = (_Float16)val;
}

__global__ __launch_bounds__(HD_) void rope_cvt_kv_kernel(
    const float* __restrict__ f, const float* __restrict__ cosT,
    const float* __restrict__ sinT, _Float16* __restrict__ out,
    int doRope, int transposeOut) {
  const int s = blockIdx.x, h = blockIdx.y, b = blockIdx.z, d = threadIdx.x;
  const float* row = f + (((size_t)(b * S_ + s)) * KV_ + h) * HD_;
  float val = row[d];
  if (doRope) {
    const float rot = (d < HD_ / 2) ? -row[d + HD_ / 2] : row[d - HD_ / 2];
    val = val * cosT[s * HD_ + d] + rot * sinT[s * HD_ + d];
  }
  const size_t idx = transposeOut
      ? ((((size_t)(b * KV_ + h)) * HD_ + d) * S_ + s)
      : ((((size_t)(b * KV_ + h)) * S_ + s) * HD_ + d);
  out[idx] = (_Float16)val;
}

__global__ __launch_bounds__(32) void attn_kernel(
    const _Float16* __restrict__ qh, const _Float16* __restrict__ kh,
    const _Float16* __restrict__ vt, float* __restrict__ attnOut) {
  __shared__ _Float16 Pl[16 * 32];
  const int qt = blockIdx.x;
  const int h = blockIdx.y;
  const int b = blockIdx.z;
  const int kvh = h / GROUP_;
  const int lane = threadIdx.x;
  const int ln = lane & 15;
  const int hl = lane >> 4;
  const int q0 = qt * 16;
  const float NEG = -1e30f;

  v16h qf[4];
  {
    const int* qr = (const int*)(qh + (((size_t)(b * H_ + h)) * S_ + q0 + ln) * HD_);
#pragma unroll
    for (int c = 0; c < 4; ++c) {
      union { int i[8]; v16h v; } u;
#pragma unroll
      for (int j = 0; j < 8; ++j)
        u.i[j] = qr[c * 16 + (j >> 2) * 8 + (j & 3) + 4 * hl];
      qf[c] = u.v;
    }
  }

  v8f acc[8];
  for (int t = 0; t < 8; ++t)
    for (int e = 0; e < 8; ++e) acc[t][e] = 0.0f;
  float mrow[8], lrow[8];
  for (int r = 0; r < 8; ++r) { mrow[r] = NEG; lrow[r] = 0.0f; }

  const _Float16* kbase = kh + ((size_t)(b * KV_ + kvh)) * S_ * HD_;
  const _Float16* vbase = vt + ((size_t)(b * KV_ + kvh)) * HD_ * S_;

  const int nTiles = qt + 1;
  for (int t0 = 0; t0 < nTiles; t0 += 2) {
    v8f sc[2];
#pragma unroll
    for (int sub = 0; sub < 2; ++sub) {
      const int kv16 = (t0 + sub) * 16;
      v8f s;
      for (int e = 0; e < 8; ++e) s[e] = 0.0f;
      const int* kr = (const int*)(kbase + (size_t)(kv16 + ln) * HD_);
#pragma unroll
      for (int c = 0; c < 4; ++c) {
        union { int i[8]; v16h v; } u;
#pragma unroll
        for (int j = 0; j < 8; ++j)
          u.i[j] = kr[c * 16 + (j >> 2) * 8 + (j & 3) + 4 * hl];
        s = __builtin_amdgcn_wmma_f32_16x16x32_f16(
            false, qf[c], false, u.v, (short)0, s, false, false);
      }
      sc[sub] = s;
    }

#pragma unroll
    for (int sub = 0; sub < 2; ++sub) {
      const int kv = (t0 + sub) * 16 + ln;
#pragma unroll
      for (int r = 0; r < 8; ++r)
        if (kv > q0 + r + 8 * hl) sc[sub][r] = NEG;
    }

    float al[8];
#pragma unroll
    for (int r = 0; r < 8; ++r) {
      float v = fmaxf(sc[0][r], sc[1][r]);
      for (int off = 8; off >= 1; off >>= 1)
        v = fmaxf(v, __shfl_xor(v, off, 32));
      const float mnew = fmaxf(mrow[r], v);
      sc[0][r] = expf(sc[0][r] - mnew);
      sc[1][r] = expf(sc[1][r] - mnew);
      float sum = sc[0][r] + sc[1][r];
      for (int off = 8; off >= 1; off >>= 1)
        sum += __shfl_xor(sum, off, 32);
      al[r] = expf(mrow[r] - mnew);
      lrow[r] = lrow[r] * al[r] + sum;
      mrow[r] = mnew;
    }
#pragma unroll
    for (int t = 0; t < 8; ++t)
#pragma unroll
      for (int r = 0; r < 8; ++r) acc[t][r] *= al[r];

#pragma unroll
    for (int sub = 0; sub < 2; ++sub)
#pragma unroll
      for (int r = 0; r < 8; ++r)
        Pl[(r + 8 * hl) * 32 + sub * 16 + ln] = (_Float16)sc[sub][r];
    __syncthreads();
    union { int i[8]; v16h v; } pu;
    {
      const int* pr = (const int*)(Pl + ln * 32);
#pragma unroll
      for (int j = 0; j < 8; ++j)
        pu.i[j] = pr[(j & 3) + 4 * hl + (j >> 2) * 8];
    }
    __syncthreads();

#pragma unroll
    for (int t = 0; t < 8; ++t) {
      union { int i[8]; v16h v; } vu;
      const int* vr = (const int*)(vbase + (size_t)(t * 16 + ln) * S_);
#pragma unroll
      for (int j = 0; j < 8; ++j)
        vu.i[j] = vr[t0 * 8 + (j >> 2) * 8 + (j & 3) + 4 * hl];
      acc[t] = __builtin_amdgcn_wmma_f32_16x16x32_f16(
          false, pu.v, false, vu.v, (short)0, acc[t], false, false);
    }
  }

#pragma unroll
  for (int t = 0; t < 8; ++t)
#pragma unroll
    for (int r = 0; r < 8; ++r) {
      const int q = q0 + r + 8 * hl;
      const int d = t * 16 + ln;
      attnOut[(((size_t)(b * S_ + q)) * H_ + h) * HD_ + d] = acc[t][r] / lrow[r];
    }
}

extern "C" void kernel_launch(void* const* d_in, const int* in_sizes, int n_in,
                              void* d_out, int out_size, void* d_ws, size_t ws_size,
                              hipStream_t stream) {
  (void)in_sizes; (void)n_in; (void)out_size; (void)ws_size;
  const float*  x    = (const float*)d_in[0];
  const float*  cosT = (const float*)d_in[1];
  const float*  sinT = (const float*)d_in[2];
  const int8_t* wq   = (const int8_t*)d_in[3];
  const float*  sq   = (const float*)d_in[4];
  const float*  bq   = (const float*)d_in[5];
  const int8_t* wk   = (const int8_t*)d_in[6];
  const float*  sk   = (const float*)d_in[7];
  const float*  bk   = (const float*)d_in[8];
  const int8_t* wv   = (const int8_t*)d_in[9];
  const float*  sv   = (const float*)d_in[10];
  const float*  bv   = (const float*)d_in[11];
  const int8_t* wo   = (const int8_t*)d_in[12];
  const float*  so   = (const float*)d_in[13];
  float* out = (float*)d_out;

  const int M = B_ * S_;
  const int DKV = KV_ * HD_;

  char* ws = (char*)d_ws;
  size_t off = 0;
  auto alloc = [&](size_t n) -> char* {
    char* p = ws + off;
    off += (n + 255) & ~(size_t)255;
    return p;
  };
  int8_t*   xq  = (int8_t*)alloc((size_t)M * D_);
  float*    sx  = (float*)alloc((size_t)M * sizeof(float));
  float*    qf  = (float*)alloc((size_t)M * D_ * 4);
  float*    kf  = (float*)alloc((size_t)M * DKV * 4);
  float*    vf  = (float*)alloc((size_t)M * DKV * 4);
  _Float16* qh  = (_Float16*)alloc((size_t)M * D_ * 2);
  _Float16* kh  = (_Float16*)alloc((size_t)M * DKV * 2);
  _Float16* vhT = (_Float16*)alloc((size_t)M * DKV * 2);

  quant_rows_kernel<<<M, 256, 0, stream>>>(x, xq, sx, D_);

  dim3 gq(D_ / 128, M / 128);
  dim3 gkv(DKV / 128, M / 128);
  gemm_i8_kernel<<<gq,  256, 0, stream>>>(xq, sx, wq, sq, bq, qf, M, D_,  D_);
  gemm_i8_kernel<<<gkv, 256, 0, stream>>>(xq, sx, wk, sk, bk, kf, M, DKV, D_);
  gemm_i8_kernel<<<gkv, 256, 0, stream>>>(xq, sx, wv, sv, bv, vf, M, DKV, D_);

  rope_cvt_q_kernel <<<dim3(S_, H_,  B_), HD_, 0, stream>>>(qf, cosT, sinT, qh);
  rope_cvt_kv_kernel<<<dim3(S_, KV_, B_), HD_, 0, stream>>>(kf, cosT, sinT, kh, 1, 0);
  rope_cvt_kv_kernel<<<dim3(S_, KV_, B_), HD_, 0, stream>>>(vf, cosT, sinT, vhT, 0, 1);

  attn_kernel<<<dim3(S_ / 16, H_, B_), 32, 0, stream>>>(qh, kh, vhT, qf);

  quant_rows_kernel<<<M, 256, 0, stream>>>(qf, xq, sx, D_);
  gemm_i8_kernel<<<gq, 256, 0, stream>>>(xq, sx, wo, so, nullptr, out, M, D_, D_);
}
